// CustomGraphNet_51874615001141
// MI455X (gfx1250) — matrix operand order repaired and checked by reading, not compiled, not run on hardware
//
#include <hip/hip_runtime.h>

// ---------------------------------------------------------------------------
// CDNA5 (gfx1250) GNN forward: bf16 WMMA GEMM pipeline, async-LDS-staged A,
// software-pipelined B fragments.
// ---------------------------------------------------------------------------

#define LATENT   128
#define NSPARSE  6000
#define NDENSE   30000
#define ESPARSE  72000
#define EDENSE   360000
#define EMULTI   (NDENSE * 3)

// Emit CDNA5 async global->LDS copies via inline asm (builtin's param types
// are not spellable from HIP source on this toolchain).
#define USE_ASYNC_LDS 1

typedef __attribute__((ext_vector_type(16))) __bf16 v16bf;
typedef __attribute__((ext_vector_type(8)))  float  v8f;

union BF16x16 { uint4 u[2]; v16bf v; };

// --- weight prep: fp32 row-major W[K][N] -> bf16 transposed+padded Wt[N][Kpad]
__global__ __launch_bounds__(256)
void prep_weight_kernel(const float* __restrict__ W, __bf16* __restrict__ Wt,
                        int Kreal, int Kpad, int N)
{
    int total = N * Kpad;
    for (int i = blockIdx.x * blockDim.x + threadIdx.x; i < total;
         i += gridDim.x * blockDim.x) {
        int n = i / Kpad;
        int k = i - n * Kpad;
        float v = (k < Kreal) ? W[(size_t)k * N + n] : 0.0f;
        Wt[i] = (__bf16)v;
    }
}

// --- input prep: fp32 [M,Din] -> bf16 [M,Kpad] zero padded
__global__ __launch_bounds__(256)
void pad_convert_kernel(const float* __restrict__ X, __bf16* __restrict__ Y,
                        long long M, int Din, int Kpad)
{
    long long total = M * (long long)Kpad;
    for (long long i = (long long)blockIdx.x * blockDim.x + threadIdx.x; i < total;
         i += (long long)gridDim.x * blockDim.x) {
        long long r = i / Kpad;
        int c = (int)(i - r * Kpad);
        Y[i] = (__bf16)((c < Din) ? X[r * Din + c] : 0.0f);
    }
}

// --- gather+concat, 16-byte granularity: dst[m] = [ nodes[idx?idx[m]:m] | sb/sf[m] ]
// K0, K1 multiples of 8 (always 128 here).
__global__ __launch_bounds__(256)
void gather_concat_kernel(const __bf16* __restrict__ nodes, const int* __restrict__ idx,
                          const __bf16* __restrict__ sb, const float* __restrict__ sf,
                          __bf16* __restrict__ dst, long long M, int K0, int K1)
{
    int c0 = K0 >> 3, c1 = K1 >> 3, cpr = c0 + c1;
    long long total = M * (long long)cpr;
    for (long long i = (long long)blockIdx.x * blockDim.x + threadIdx.x; i < total;
         i += (long long)gridDim.x * blockDim.x) {
        long long r = i / cpr;
        int c = (int)(i - r * cpr);
        uint4* d = (uint4*)(dst + r * (K0 + K1)) + c;
        if (c < c0) {
            long long j = idx ? (long long)idx[r] : r;
            *d = *((const uint4*)(nodes + j * K0) + c);
        } else {
            int e = (c - c0) << 3;
            if (sb) {
                *d = *((const uint4*)(sb + r * K1 + e));
            } else {
                const float* s = sf + r * K1 + e;
                __bf16 t[8];
#pragma unroll
                for (int q = 0; q < 8; ++q) t[q] = (__bf16)s[q];
                *d = *(const uint4*)t;
            }
        }
    }
}

// --- segment_sum via f32 atomics
__global__ __launch_bounds__(256)
void scatter_add_kernel(const __bf16* __restrict__ edges, const int* __restrict__ dsti,
                        float* __restrict__ agg, long long E)
{
    long long total = E * (long long)LATENT;
    for (long long i = (long long)blockIdx.x * blockDim.x + threadIdx.x; i < total;
         i += (long long)gridDim.x * blockDim.x) {
        long long e = i >> 7;
        int c = (int)(i & 127);
        atomicAdd(&agg[(long long)dsti[e] * LATENT + c], (float)edges[i]);
    }
}

// --- B fragment load: 32 contiguous bf16 of Wt row (two b128)
__device__ __forceinline__ void load_b_frag(const __bf16* brow, int k, int kboff,
                                            BF16x16& t)
{
    t.u[0] = *(const uint4*)(brow + k + kboff);
    t.u[1] = *(const uint4*)(brow + k + kboff + 16);
}

// --- one K-step: 4 A fragments from LDS (batched), 4 WMMAs sharing B
__device__ __forceinline__ void k_step(const __bf16* arow, int k, int kaoff, int Kpad,
                                       const BF16x16& tb, v8f acc[4])
{
    const __bf16* ap = arow + k + kaoff;
    BF16x16 a0, a1, a2, a3;
    a0.u[0] = *(const uint4*)(ap);             a0.u[1] = *(const uint4*)(ap + 16);
    a1.u[0] = *(const uint4*)(ap + 16 * Kpad); a1.u[1] = *(const uint4*)(ap + 16 * Kpad + 16);
    a2.u[0] = *(const uint4*)(ap + 32 * Kpad); a2.u[1] = *(const uint4*)(ap + 32 * Kpad + 16);
    a3.u[0] = *(const uint4*)(ap + 48 * Kpad); a3.u[1] = *(const uint4*)(ap + 48 * Kpad + 16);
    acc[0] = __builtin_amdgcn_wmma_f32_16x16x32_bf16(false, a0.v, false, tb.v, (short)0, acc[0], false, false);
    acc[1] = __builtin_amdgcn_wmma_f32_16x16x32_bf16(false, a1.v, false, tb.v, (short)0, acc[1], false, false);
    acc[2] = __builtin_amdgcn_wmma_f32_16x16x32_bf16(false, a2.v, false, tb.v, (short)0, acc[2], false, false);
    acc[3] = __builtin_amdgcn_wmma_f32_16x16x32_bf16(false, a3.v, false, tb.v, (short)0, acc[3], false, false);
}

// --- core GEMM:  out[M,N] = act(A[M,Kpad] @ W + bias) (+ residual)
// Block tile: 64 rows x N cols, 8 waves. A tile staged in LDS with
// global_load_async_to_lds_b128 (ASYNCcnt). Wave w = N-tile w; 4 M-tiles per
// wave; B fragments double-buffered one K-step ahead so global latency is
// hidden behind 8 ds_loads + 4 WMMAs.
__global__ __launch_bounds__(256)
void wmma_linear_kernel(const __bf16* __restrict__ A, int lda,
                        const __bf16* __restrict__ Wt, const float* __restrict__ bias,
                        const __bf16* __restrict__ residual,
                        __bf16* __restrict__ outB, float* __restrict__ outF,
                        int M, int Kpad, int N, int relu)
{
    extern __shared__ __bf16 lds[];                 // [64][Kpad]
    const int tid  = threadIdx.x;
    const int lane = tid & 31;
    const int wave = tid >> 5;                      // 0..7 = N-tile
    const int mblk = blockIdx.x * 64;

    // ---- stage A tile [64 x Kpad] bf16 into LDS, 16B chunks ----
    const int cpr = Kpad >> 3;                      // 16B chunks per row
    for (int c = tid; c < 64 * cpr; c += 256) {
        int r   = c / cpr;
        int col = (c - r * cpr) << 3;
        int rg  = mblk + r; if (rg >= M) rg = M - 1;  // clamp; stores guarded later
        const __bf16* src = A + (size_t)rg * lda + col;
        __bf16*       dst = lds + r * Kpad + col;
#if USE_ASYNC_LDS
        unsigned ldsoff = (unsigned)((const char*)dst - (const char*)lds);
        asm volatile("global_load_async_to_lds_b128 %0, %1, off"
                     :: "v"(ldsoff), "v"(src) : "memory");
#else
        *(uint4*)dst = *(const uint4*)src;
#endif
    }
#if USE_ASYNC_LDS
    asm volatile("s_wait_asynccnt 0" ::: "memory");
#endif
    __syncthreads();

    const int ntile = wave;
    if (ntile * 16 >= N) return;                    // wave-uniform; after barrier

    int n   = ntile * 16 + (lane & 15);
    int ncl = (n < N) ? n : (N - 1);
    const __bf16* brow = Wt + (size_t)ncl * Kpad;
    const __bf16* arow = lds + (lane & 15) * Kpad;
    const int kaoff = (lane >> 4) << 3;             // 0 or 8   (A K-halves)
    const int kboff = (lane >> 4) << 3;             // 0 or 8   (B K-halves)

    v8f acc[4] = {};
    BF16x16 t0, t1;
    load_b_frag(brow, 0, kboff, t0);
    int k = 0;
    for (; k + 64 <= Kpad; k += 64) {               // 2 K-steps per iter, B pipelined
        load_b_frag(brow, k + 32, kboff, t1);
        k_step(arow, k, kaoff, Kpad, t0, acc);
        if (k + 64 < Kpad) load_b_frag(brow, k + 64, kboff, t0);
        k_step(arow, k + 32, kaoff, Kpad, t1, acc);
    }
    if (k < Kpad)                                   // odd tail (Kpad == 32)
        k_step(arow, k, kaoff, Kpad, t0, acc);

    float bv = (n < N) ? bias[n] : 0.0f;
    const int mo = (lane >> 4) << 3;                // +8 rows for upper half-wave
#pragma unroll
    for (int mt = 0; mt < 4; ++mt) {
#pragma unroll
        for (int r = 0; r < 8; ++r) {
            int m = mblk + mt * 16 + r + mo;
            if (m < M && n < N) {
                float v = acc[mt][r] + bv;
                if (relu && v < 0.0f) v = 0.0f;
                size_t o = (size_t)m * N + n;
                if (residual) v += (float)residual[o];
                if (outB) outB[o] = (__bf16)v;
                if (outF) outF[o] = v;
            }
        }
    }
}

// ---------------------------------------------------------------------------
// Host orchestration
// ---------------------------------------------------------------------------

extern "C" void kernel_launch(void* const* d_in, const int* in_sizes, int n_in,
                              void* d_out, int out_size, void* d_ws, size_t ws_size,
                              hipStream_t stream)
{
    (void)in_sizes; (void)n_in; (void)out_size; (void)ws_size;

    const float* x_ns = (const float*)d_in[0];   // [NSPARSE,16]
    const float* x_nd = (const float*)d_in[1];   // [NDENSE,13]
    const float* x_es = (const float*)d_in[2];   // [ESPARSE,9]
    const float* x_ed = (const float*)d_in[3];   // [EDENSE,9]
    const float* x_em = (const float*)d_in[4];   // [EMULTI,12]
    const int*   ei_s = (const int*)d_in[5];     // [2,ESPARSE]
    const int*   ei_d = (const int*)d_in[6];     // [2,EDENSE]
    const int*   ei_m = (const int*)d_in[7];     // [2,EMULTI]

    struct Lin  { const float* w; const float* b; int K; int N; };
    struct Fnet { Lin fin, ra, rb, fout; };

    // params flattened in jax sorted-key pytree order, 8 leaves per fnet:
    // fin.b, fin.w, fout.b, fout.w, res.a.b, res.a.w, res.b.b, res.b.w
    int p = 8;
    auto take = [&](int K, int Nout) -> Fnet {
        Fnet f;
        f.fin.b  = (const float*)d_in[p + 0];  f.fin.w  = (const float*)d_in[p + 1];
        f.fout.b = (const float*)d_in[p + 2];  f.fout.w = (const float*)d_in[p + 3];
        f.ra.b   = (const float*)d_in[p + 4];  f.ra.w   = (const float*)d_in[p + 5];
        f.rb.b   = (const float*)d_in[p + 6];  f.rb.w   = (const float*)d_in[p + 7];
        p += 8;
        f.fin.K = K;      f.fin.N = LATENT;
        f.ra.K  = LATENT; f.ra.N  = LATENT;
        f.rb.K  = LATENT; f.rb.N  = LATENT;
        f.fout.K = LATENT; f.fout.N = Nout;
        return f;
    };
    // top-level sorted: decoder, dense_proc, down, enc_edge_dense, enc_edge_multi,
    // enc_edge_sparse, enc_node_dense, enc_node_sparse, sparse_proc
    Fnet decoder = take(LATENT, 3);
    Fnet dproc[4][2];
    for (int i = 0; i < 4; ++i) { dproc[i][0] = take(256, 128); dproc[i][1] = take(256, 128); }
    Fnet down_inner = take(384, 128);   // down.dec_inner ("dec_inner" < "edge")
    Fnet down_edge  = take(256, 128);
    Fnet enc_ed = take(9,  128);
    Fnet enc_em = take(12, 128);
    Fnet enc_es = take(9,  128);
    Fnet enc_nd = take(13, 128);
    Fnet enc_ns = take(16, 128);
    Fnet sproc[4][2];
    for (int i = 0; i < 4; ++i) { sproc[i][0] = take(256, 128); sproc[i][1] = take(256, 128); }

    // ------------------ workspace layout ------------------
    char* base = (char*)d_ws; size_t off = 0;
    auto alloc = [&](size_t bytes) -> void* {
        void* r = base + off; off += (bytes + 255) & ~(size_t)255; return r;
    };
    __bf16* Wt   = (__bf16*)alloc((size_t)384 * 128 * 2);          // transposed weight
    __bf16* xpad = (__bf16*)alloc((size_t)EDENSE * 32 * 2);        // padded encoder input
    __bf16* ns   = (__bf16*)alloc((size_t)NSPARSE * LATENT * 2);
    __bf16* nd   = (__bf16*)alloc((size_t)NDENSE  * LATENT * 2);
    __bf16* esb  = (__bf16*)alloc((size_t)ESPARSE * LATENT * 2);
    __bf16* edb  = (__bf16*)alloc((size_t)EDENSE  * LATENT * 2);
    __bf16* emb  = (__bf16*)alloc((size_t)EMULTI  * LATENT * 2);
    __bf16* h    = (__bf16*)alloc((size_t)EDENSE  * LATENT * 2);
    __bf16* y    = (__bf16*)alloc((size_t)EDENSE  * LATENT * 2);
    __bf16* cc   = (__bf16*)alloc((size_t)EDENSE  * 256 * 2);      // concat / new_edge
    __bf16* le   = (__bf16*)alloc((size_t)EMULTI  * LATENT * 2);   // link edges
    float*  agg  = (float*) alloc((size_t)NDENSE  * LATENT * 4);

    // ------------------ launch helpers ------------------
    auto egrid = [](long long total) {
        long long g = (total + 255) / 256; if (g > 16384) g = 16384;
        return dim3((unsigned)g);
    };
    auto linear = [&](const __bf16* A, int lda, int Kpad, const Lin& L, int M,
                      const __bf16* res, __bf16* oB, float* oF, int relu) {
        prep_weight_kernel<<<egrid((long long)L.N * Kpad), 256, 0, stream>>>(
            L.w, Wt, L.K, Kpad, L.N);
        size_t ldsb = (size_t)64 * Kpad * 2;        // A-tile staging
        wmma_linear_kernel<<<dim3((M + 63) / 64), 256, ldsb, stream>>>(
            A, lda, Wt, L.b, res, oB, oF, M, Kpad, L.N, relu);
    };
    // ForwardNet: relu(fin) -> h; y=relu(h@a); h += relu(y@b); out = h@fout + res
    auto fnet = [&](const Fnet& f, const __bf16* A, int lda, int Kpad, int M,
                    const __bf16* res, __bf16* oB, float* oF) {
        linear(A, lda, Kpad, f.fin, M, nullptr, h, nullptr, 1);
        linear(h, 128, 128, f.ra, M, nullptr, y, nullptr, 1);
        linear(y, 128, 128, f.rb, M, h, h, nullptr, 1);   // h = h + relu(...)
        linear(h, 128, 128, f.fout, M, res, oB, oF, 0);
    };
    auto gc = [&](const __bf16* nodes, const int* idx, const __bf16* sb, const float* sf,
                  __bf16* dst, long long M, int K0, int K1) {
        gather_concat_kernel<<<egrid(M * ((K0 + K1) >> 3)), 256, 0, stream>>>(
            nodes, idx, sb, sf, dst, M, K0, K1);
    };
    auto pad = [&](const float* X, long long M, int Din) {
        pad_convert_kernel<<<egrid(M * 32), 256, 0, stream>>>(X, xpad, M, Din, 32);
    };
    auto process = [&](const Fnet& fe, const Fnet& fn, __bf16* nl, const __bf16* el,
                       const int* src, const int* dst, int Nn, long long E) {
        gc(nl, src, el, nullptr, cc, E, 128, 128);                 // [nl[src] | edge]
        fnet(fe, cc, 256, 256, (int)E, nullptr, cc, nullptr);      // new_edge -> cc
        (void)hipMemsetAsync(agg, 0, (size_t)Nn * LATENT * 4, stream);
        scatter_add_kernel<<<egrid(E * LATENT), 256, 0, stream>>>(cc, dst, agg, E);
        gc(nl, nullptr, nullptr, agg, cc, Nn, 128, 128);           // [nl | agg]
        fnet(fn, cc, 256, 256, Nn, nl, nl, nullptr);               // nl += fnet(...)
    };

    // ------------------ pipeline ------------------
    // Encoders (fin padded to K=32)
    pad(x_ns, NSPARSE, 16); fnet(enc_ns, xpad, 32, 32, NSPARSE, nullptr, ns,  nullptr);
    pad(x_nd, NDENSE,  13); fnet(enc_nd, xpad, 32, 32, NDENSE,  nullptr, nd,  nullptr);
    pad(x_es, ESPARSE,  9); fnet(enc_es, xpad, 32, 32, ESPARSE, nullptr, esb, nullptr);
    pad(x_ed, EDENSE,   9); fnet(enc_ed, xpad, 32, 32, EDENSE,  nullptr, edb, nullptr);
    pad(x_em, EMULTI,  12); fnet(enc_em, xpad, 32, 32, EMULTI,  nullptr, emb, nullptr);

    // 4 sparse message passes
    for (int i = 0; i < 4; ++i)
        process(sproc[i][0], sproc[i][1], ns, esb, ei_s, ei_s + ESPARSE, NSPARSE, ESPARSE);

    // Down link: edge MLP over [ns[src] | multi edge], then inner decoder over
    // the 3 grouped link edges (le viewed as [NDENSE, 384]) with residual into nd.
    gc(ns, ei_m, emb, nullptr, cc, EMULTI, 128, 128);
    fnet(down_edge,  cc, 256, 256, EMULTI, nullptr, le, nullptr);
    fnet(down_inner, le, 384, 384, NDENSE, nd, nd, nullptr);

    // 4 dense message passes
    for (int i = 0; i < 4; ++i)
        process(dproc[i][0], dproc[i][1], nd, edb, ei_d, ei_d + EDENSE, NDENSE, EDENSE);

    // Decoder: [NDENSE,128] -> [NDENSE,3] f32 straight into d_out
    fnet(decoder, nd, 128, 128, NDENSE, nullptr, nullptr, (float*)d_out);
}
